// GroupedExperts_50921132261883
// MI455X (gfx1250) — matrix operand order repaired and checked by reading, not compiled, not run on hardware
//
#include <hip/hip_runtime.h>

// ---------------------------------------------------------------------------
// Grouped SwiGLU MoE FFN for MI455X (gfx1250, wave32, WMMA bf16 16x16x32).
//   out_e = (silu(x_e @ w1_e^T) * (x_e @ w3_e^T)) @ w2_e^T
// x:[32768,1024] f32, w1/w3:[64,512,1024] f32, w2:[64,1024,512] f32.
//
// Pass 1: convert w1/w3/w2 fp32 -> bf16 into d_ws (memory-bound, no reuse).
// Pass 2: fused grouped GEMM; weights stream as bf16 (global_load_b128,
//         zero conversion VALU in the K-loop), x staged once in LDS as bf16.
// Wave tiling: 8 waves = 2(M: 32 rows, 2 A-frags) x 4(N: 32 cols, 2 tiles).
// ---------------------------------------------------------------------------

typedef __attribute__((ext_vector_type(16))) __bf16 v16bf;
typedef __attribute__((ext_vector_type(8)))  float  v8f;

union FragAB {
    v16bf        v;
    unsigned int u[8];
    uint4        q[2];
};

// Pack two f32 -> bf16x2 (low = a): round-half-up bias + one v_perm_b32.
__device__ __forceinline__ unsigned int pk_bf16(float a, float b) {
    union { float f; unsigned int u; } ca, cb;
    ca.f = a; cb.f = b;
    return __builtin_amdgcn_perm(cb.u + 0x8000u, ca.u + 0x8000u, 0x07060302u);
}

__device__ __forceinline__ unsigned short bf16_1(float a) {
    union { float f; unsigned int u; } c;
    c.f = a;
    return (unsigned short)((c.u + 0x8000u) >> 16);
}

// 32x16 bf16 B-fragment from fp32 weights (fallback path): 4x float4 + 8 packs.
__device__ __forceinline__ void load_b_f32(const float* __restrict__ p, FragAB& b) {
    const float4* q = reinterpret_cast<const float4*>(p);
    float4 f0 = q[0], f1 = q[1], f2 = q[2], f3 = q[3];
    b.u[0] = pk_bf16(f0.x, f0.y); b.u[1] = pk_bf16(f0.z, f0.w);
    b.u[2] = pk_bf16(f1.x, f1.y); b.u[3] = pk_bf16(f1.z, f1.w);
    b.u[4] = pk_bf16(f2.x, f2.y); b.u[5] = pk_bf16(f2.z, f2.w);
    b.u[6] = pk_bf16(f3.x, f3.y); b.u[7] = pk_bf16(f3.z, f3.w);
}

// B fragment: lane holds column n=(lane&15); lanes 0-15 carry K=kk..kk+15,
// lanes 16-31 carry K=kk+16..kk+31 (contiguous K order, same as A). idx in elements.
template <bool WBF16>
__device__ __forceinline__ void load_b(const void* __restrict__ wbase, size_t idx, FragAB& b) {
    if constexpr (WBF16) {
        const uint4* q =
            reinterpret_cast<const uint4*>(reinterpret_cast<const unsigned short*>(wbase) + idx);
        b.q[0] = q[0];             // 2x global_load_b128, no VALU
        b.q[1] = q[1];
    } else {
        load_b_f32(reinterpret_cast<const float*>(wbase) + idx, b);
    }
}

#define WMMA_BF16(A, B, C) \
    __builtin_amdgcn_wmma_f32_16x16x32_bf16(false, (A), false, (B), (short)0, (C), false, false)

constexpr int TPB    = 256;   // 8 waves
constexpr int DIM    = 1024;
constexpr int HIDDEN = 512;
constexpr int MT     = 64;    // token rows per block

// ---- Pass 1: fp32 -> bf16 weight conversion (grid-stride-free, 1:1) -------
__global__ void convert_w_bf16(const float* __restrict__ src,
                               unsigned short* __restrict__ dst, long n4) {
    const long i = (long)blockIdx.x * blockDim.x + threadIdx.x;
    if (i >= n4) return;
    const float4 f = reinterpret_cast<const float4*>(src)[i];
    reinterpret_cast<uint2*>(dst)[i] = make_uint2(pk_bf16(f.x, f.y), pk_bf16(f.z, f.w));
}

// ---- Pass 2: fused grouped SwiGLU ------------------------------------------
template <bool WBF16>
__launch_bounds__(TPB, 1)
__global__ void moe_swiglu_wmma(const float* __restrict__ x,
                                const int*   __restrict__ counts,
                                const void*  __restrict__ w1,
                                const void*  __restrict__ w2,
                                const void*  __restrict__ w3,
                                float*       __restrict__ out,
                                int n_tokens) {
    __shared__ unsigned short x_lds[MT * DIM];     // 128 KB bf16 x-tile
    __shared__ unsigned short h_lds[MT * HIDDEN];  //  64 KB bf16 h-tile

    const int e    = blockIdx.x >> 3;   // expert
    const int tile = blockIdx.x & 7;    // 64-row tile within expert

    int base = 0;
    for (int i = 0; i < e; ++i) base += counts[i];
    const int cnt = counts[e];
    const int m0  = tile * MT;
    if (m0 >= cnt) return;              // uniform per block (before barriers)
    const int rows_here = min(MT, cnt - m0);

    const int tid  = threadIdx.x;
    const int lane = tid & 31;
    const int wave = tid >> 5;
    const int wm   = wave & 1;          // M group: rows wm*32 .. +31
    const int wn   = wave >> 1;         // N group: 0..3
    const int half = lane >> 4;
    const int l16  = lane & 15;

    // ---- Phase 0: stage x tile (fp32 -> bf16) into LDS ---------------------
    for (int i = tid; i < (MT * DIM) / 4; i += TPB) {
        const int flat = i * 4;
        const int r = flat >> 10;           // / DIM
        const int c = flat & (DIM - 1);
        long gr = (long)base + m0 + r;
        if (gr >= n_tokens) gr = n_tokens - 1;   // clamp (partial last tile)
        const float4 f = *reinterpret_cast<const float4*>(x + gr * DIM + c);
        *reinterpret_cast<uint2*>(&x_lds[r * DIM + c]) =
            make_uint2(pk_bf16(f.x, f.y), pk_bf16(f.z, f.w));
    }
    __syncthreads();

    // A fragment addressing (contiguous K order, matching B):
    //   lane<16 : row=lane,    v0..7 = K 0..15
    //   lane>=16: row=lane-16, v0..7 = K 16..31
    const int arow0 = wm * 32 + l16;
    const int arow1 = arow0 + 16;
    const int akoff = half * 16;
    const int bkoff = half * 16;

    // ---- Phase 1: h = silu(x @ w1^T) * (x @ w3^T) --------------------------
    const size_t w13off = (size_t)e * HIDDEN * DIM;

    for (int c = 0; c < HIDDEN / 128; ++c) {       // 4 chunks of N=128
        const int n0 = c * 128 + wn * 32;
        v8f acc1[2][2], acc3[2][2];
        #pragma unroll
        for (int f = 0; f < 2; ++f)
            #pragma unroll
            for (int t = 0; t < 2; ++t) { acc1[f][t] = {}; acc3[f][t] = {}; }

        for (int kk = 0; kk < DIM; kk += 32) {
            FragAB a0, a1, b;
            a0.q[0] = *reinterpret_cast<const uint4*>(&x_lds[arow0 * DIM + kk + akoff]);
            a0.q[1] = *reinterpret_cast<const uint4*>(&x_lds[arow0 * DIM + kk + akoff + 8]);
            a1.q[0] = *reinterpret_cast<const uint4*>(&x_lds[arow1 * DIM + kk + akoff]);
            a1.q[1] = *reinterpret_cast<const uint4*>(&x_lds[arow1 * DIM + kk + akoff + 8]);
            #pragma unroll
            for (int t = 0; t < 2; ++t) {
                const size_t rowo = w13off + (size_t)(n0 + t * 16 + l16) * DIM + kk + bkoff;
                load_b<WBF16>(w1, rowo, b);
                acc1[0][t] = WMMA_BF16(a0.v, b.v, acc1[0][t]);
                acc1[1][t] = WMMA_BF16(a1.v, b.v, acc1[1][t]);
                load_b<WBF16>(w3, rowo, b);
                acc3[0][t] = WMMA_BF16(a0.v, b.v, acc3[0][t]);
                acc3[1][t] = WMMA_BF16(a1.v, b.v, acc3[1][t]);
            }
        }
        // C/D layout: VGPR i -> row (rbase + i), col (lane&15)
        #pragma unroll
        for (int f = 0; f < 2; ++f) {
            const int rbase = wm * 32 + f * 16 + half * 8;
            #pragma unroll
            for (int t = 0; t < 2; ++t) {
                const int col = n0 + t * 16 + l16;
                #pragma unroll
                for (int i = 0; i < 8; ++i) {
                    const float g = acc1[f][t][i];
                    const float s = g / (1.0f + __expf(-g));
                    h_lds[(rbase + i) * HIDDEN + col] = bf16_1(s * acc3[f][t][i]);
                }
            }
        }
    }
    __syncthreads();

    // ---- Phase 2: out = h @ w2^T ------------------------------------------
    const size_t w2off = (size_t)e * DIM * HIDDEN;

    for (int c = 0; c < DIM / 128; ++c) {          // 8 chunks of N=128
        const int n0 = c * 128 + wn * 32;
        v8f acc[2][2];
        #pragma unroll
        for (int f = 0; f < 2; ++f) { acc[f][0] = {}; acc[f][1] = {}; }

        for (int kk = 0; kk < HIDDEN; kk += 32) {
            FragAB a0, a1, b;
            a0.q[0] = *reinterpret_cast<const uint4*>(&h_lds[arow0 * HIDDEN + kk + akoff]);
            a0.q[1] = *reinterpret_cast<const uint4*>(&h_lds[arow0 * HIDDEN + kk + akoff + 8]);
            a1.q[0] = *reinterpret_cast<const uint4*>(&h_lds[arow1 * HIDDEN + kk + akoff]);
            a1.q[1] = *reinterpret_cast<const uint4*>(&h_lds[arow1 * HIDDEN + kk + akoff + 8]);
            #pragma unroll
            for (int t = 0; t < 2; ++t) {
                load_b<WBF16>(w2, w2off + (size_t)(n0 + t * 16 + l16) * HIDDEN + kk + bkoff, b);
                acc[0][t] = WMMA_BF16(a0.v, b.v, acc[0][t]);
                acc[1][t] = WMMA_BF16(a1.v, b.v, acc[1][t]);
            }
        }
        #pragma unroll
        for (int f = 0; f < 2; ++f) {
            const int rbase = wm * 32 + f * 16 + half * 8;
            #pragma unroll
            for (int t = 0; t < 2; ++t) {
                const int col = n0 + t * 16 + l16;
                #pragma unroll
                for (int i = 0; i < 8; ++i) {
                    const int rl = rbase + i;
                    if (rl < rows_here)
                        out[(size_t)(base + m0 + rl) * DIM + col] = acc[f][t][i];
                }
            }
        }
    }
}

extern "C" void kernel_launch(void* const* d_in, const int* in_sizes, int n_in,
                              void* d_out, int out_size, void* d_ws, size_t ws_size,
                              hipStream_t stream) {
    const float* x      = (const float*)d_in[0];
    const int*   counts = (const int*)  d_in[1];
    const float* w1     = (const float*)d_in[2];
    const float* w2     = (const float*)d_in[3];
    const float* w3     = (const float*)d_in[4];
    float*       out    = (float*)d_out;
    const int n_tokens  = in_sizes[0] / DIM;

    const size_t wElems = (size_t)64 * HIDDEN * DIM;           // per tensor
    const size_t needed = 3 * wElems * sizeof(unsigned short); // 192 MB

    dim3 grid(64 * 8);   // 64 experts x 8 row-tiles of 64 tokens

    if (ws_size >= needed) {
        unsigned short* w1b = (unsigned short*)d_ws;
        unsigned short* w3b = w1b + wElems;
        unsigned short* w2b = w3b + wElems;
        const long n4 = (long)(wElems / 4);
        const int cblocks = (int)((n4 + TPB - 1) / TPB);
        convert_w_bf16<<<cblocks, TPB, 0, stream>>>(w1, w1b, n4);
        convert_w_bf16<<<cblocks, TPB, 0, stream>>>(w3, w3b, n4);
        convert_w_bf16<<<cblocks, TPB, 0, stream>>>(w2, w2b, n4);
        moe_swiglu_wmma<true><<<grid, TPB, 0, stream>>>(x, counts, w1b, w2b, w3b, out, n_tokens);
    } else {
        moe_swiglu_wmma<false><<<grid, TPB, 0, stream>>>(x, counts, w1, w2, w3, out, n_tokens);
    }
}
